// SynthesisNetwork_56453050139459
// MI455X (gfx1250) — matrix operand order repaired and checked by reading, not compiled, not run on hardware
//
#include <hip/hip_runtime.h>
#include <hip/hip_bf16.h>

typedef _Float16 v16h __attribute__((ext_vector_type(16)));
typedef _Float16 v8h  __attribute__((ext_vector_type(8)));
typedef float    v8f  __attribute__((ext_vector_type(8)));

#define NPLANES 3
#define CH      32
#define HRES    256
#define WRES    256
#define HID     64
#define OUTC    33
#define OUTP    48      // OUTC padded to 3 x 16 WMMA column tiles
#define WDIM    512
#define PTS     256     // points per block (8 waves x 32 rows)
#define EPSV    1e-8f
#define CLAMPV  256.0f
#define SLOPE   0.2f
#define ACT_GAIN 1.4142135623730951f

// ---------------------------------------------------------------------------
// Kernel 0: fold style modulation + demodulation into one f16 weight matrix
// per (batch, layer):  Weff[o][j] = W[o][j] * style[j] * rsqrt(sum_j (W*style)^2)
// ---------------------------------------------------------------------------
__global__ __launch_bounds__(64)
void modw_kernel(const float* __restrict__ wlat,   // [N, 512]
                 const float* __restrict__ Wm,     // [co, ci]
                 const float* __restrict__ Ws,     // [ci, 512]
                 const float* __restrict__ bs,     // [ci]
                 _Float16*    __restrict__ weff,   // [N, co_pad, ci]
                 int ci, int co, int co_pad)
{
    __shared__ float style[64];
    __shared__ float dec[64];
    const int n   = blockIdx.x;
    const int tid = threadIdx.x;
    const float wscale = 0.044194173824159216f;    // 1/sqrt(512)

    if (tid < ci) {
        float acc = 0.f;
        const float* wp = wlat + (size_t)n * WDIM;
        const float* sp = Ws   + (size_t)tid * WDIM;
        for (int k = 0; k < WDIM; ++k) acc += wp[k] * sp[k];
        style[tid] = acc * wscale + bs[tid];
    }
    __syncthreads();
    if (tid < co) {
        float s = 0.f;
        for (int j = 0; j < ci; ++j) {
            float t = Wm[tid * ci + j] * style[j];
            s += t * t;
        }
        dec[tid] = rsqrtf(s + EPSV);
    }
    __syncthreads();
    for (int idx = tid; idx < co_pad * ci; idx += 64) {
        int o = idx / ci, j = idx - o * ci;
        float v = (o < co) ? Wm[o * ci + j] * style[j] * dec[o] : 0.f;
        weff[(size_t)n * co_pad * ci + idx] = (_Float16)v;
    }
}

// ---------------------------------------------------------------------------
// WMMA fragment helpers (CDNA5 wave32 layouts, CDNA5 ISA §7.12.2)
// ---------------------------------------------------------------------------
__device__ __forceinline__ v8f wmma_f16(v16h a, v16h b, v8f c) {
    // (neg_a, A, neg_b, B, c_mod, C, reuse_a, reuse_b)
    return __builtin_amdgcn_wmma_f32_16x16x32_f16(false, a, false, b,
                                                  (short)0, c, false, false);
}

// A (16x32 f16): lane<16 -> row=lane, halves[0..7]=K0..7, halves[8..15]=K16..23
//                lane>=16 -> row=lane-16, K8..15 and K24..31
__device__ __forceinline__ v16h loadA(const _Float16* buf, int row_stride,
                                      int rowBase, int kstep, int lane)
{
    const int lr = lane & 15, hi = lane >> 4;
    const _Float16* p = buf + (size_t)(rowBase + lr) * row_stride + kstep * 32 + hi * 8;
    v8h lo = *(const v8h*)(p);
    v8h hh = *(const v8h*)(p + 16);
    v16h a;
#pragma unroll
    for (int i = 0; i < 8; ++i) { a[i] = lo[i]; a[i + 8] = hh[i]; }
    return a;
}

// B (32x16 f16): lane<16 -> col=lane, K0..7,K16..23; lane>=16 -> col=lane-16, K8..15,K24..31
// weff stored row-major [col][ci] -> two runs of 8 consecutive halves, 16 apart, per lane.
__device__ __forceinline__ v16h loadB(const _Float16* wmat, int ci,
                                      int colBase, int kstep, int lane)
{
    const int col = colBase + (lane & 15);
    const int kb  = kstep * 32 + ((lane >> 4) << 3);
    const _Float16* p = wmat + (size_t)col * ci + kb;
    v8h lo = *(const v8h*)(p);
    v8h hh = *(const v8h*)(p + 16);
    v16h b;
#pragma unroll
    for (int i = 0; i < 8; ++i) { b[i] = lo[i]; b[i + 8] = hh[i]; }
    return b;
}

// D (16x16 f32): vgpr r -> M = r + 8*(lane>=16), N = lane&15.
// bias + lrelu(gain) + clamp epilogue, store f16 row-major to LDS.
__device__ __forceinline__ void store_act(v8f d, _Float16* buf, int row_stride,
                                          int rowBase, int colBase,
                                          const float* __restrict__ bias,
                                          bool act, int lane)
{
    const int col = colBase + (lane & 15);
    const float bv = bias[col];
    const int r0 = rowBase + ((lane >> 4) << 3);
#pragma unroll
    for (int r = 0; r < 8; ++r) {
        float v = d[r] + bv;
        if (act) v = (v >= 0.f ? v : SLOPE * v) * ACT_GAIN;
        v = fminf(fmaxf(v, -CLAMPV), CLAMPV);
        buf[(size_t)(r0 + r) * row_stride + col] = (_Float16)v;
    }
}

// ---------------------------------------------------------------------------
// Main kernel: triplane bilinear gather (1 pt / thread) + WMMA MLP
// ---------------------------------------------------------------------------
__global__ __launch_bounds__(PTS)
void triplane_mlp_kernel(const float* __restrict__ planes,  // [N,3,32,256,256]
                         const float* __restrict__ coords,  // [N,M,3]
                         const int*   __restrict__ b_idx,   // [N]
                         const _Float16* __restrict__ weff0,
                         const _Float16* __restrict__ weff1,
                         const _Float16* __restrict__ weff2,
                         const float* __restrict__ b0,
                         const float* __restrict__ b1,
                         const float* __restrict__ b2,
                         float* __restrict__ out,            // [N,M,33]
                         int M)
{
    __shared__ _Float16 sX [PTS * CH];    // 16 KB  sampled features
    __shared__ _Float16 sH1[PTS * HID];   // 32 KB  hidden 1
    __shared__ _Float16 sH2[PTS * HID];   // 32 KB  hidden 2

    const int n    = blockIdx.y;
    const int tid  = threadIdx.x;
    const int lane = tid & 31;
    const int wave = tid >> 5;
    const int m0   = blockIdx.x * PTS;

    const _Float16* w0 = weff0 + (size_t)n * HID  * CH;
    const _Float16* w1 = weff1 + (size_t)n * HID  * HID;
    const _Float16* w2 = weff2 + (size_t)n * OUTP * HID;
    __builtin_prefetch(w0, 0, 0);
    __builtin_prefetch(w1, 0, 0);
    __builtin_prefetch(w2, 0, 0);

    // ---- phase 1: triplane bilinear gather, one point per thread ----
    {
        const int m  = m0 + tid;
        const int mc = (m < M) ? m : (M - 1);
        const float* c3 = coords + ((size_t)n * M + mc) * 3;
        const float cx = c3[0] * 2.f, cy = c3[1] * 2.f, cz = c3[2] * 2.f;
        const float* fb = planes + (size_t)b_idx[n] * NPLANES * CH * HRES * WRES;
        float acc[CH];
#pragma unroll
        for (int ch = 0; ch < CH; ++ch) acc[ch] = 0.f;

#pragma unroll
        for (int pl = 0; pl < NPLANES; ++pl) {
            // plane projections: (cx,cy), (cx,cz), (cz,cy)
            const float px = (pl == 2) ? cz : cx;
            const float py = (pl == 1) ? cz : cy;
            // grid_sample: align_corners=False, padding zeros
            const float fx = (px + 1.f) * (WRES * 0.5f) - 0.5f;
            const float fy = (py + 1.f) * (HRES * 0.5f) - 0.5f;
            const float x0f = floorf(fx), y0f = floorf(fy);
            const float wx = fx - x0f, wy = fy - y0f;
            const int x0 = (int)x0f, y0 = (int)y0f;
            const int x1 = x0 + 1,  y1 = y0 + 1;
            const float vx0 = (x0 >= 0 && x0 < WRES) ? 1.f : 0.f;
            const float vx1 = (x1 >= 0 && x1 < WRES) ? 1.f : 0.f;
            const float vy0 = (y0 >= 0 && y0 < HRES) ? 1.f : 0.f;
            const float vy1 = (y1 >= 0 && y1 < HRES) ? 1.f : 0.f;
            const float w00 = (1.f - wx) * (1.f - wy) * vx0 * vy0;
            const float w10 = wx * (1.f - wy) * vx1 * vy0;
            const float w01 = (1.f - wx) * wy * vx0 * vy1;
            const float w11 = wx * wy * vx1 * vy1;
            const int xc0 = min(max(x0, 0), WRES - 1), xc1 = min(max(x1, 0), WRES - 1);
            const int yc0 = min(max(y0, 0), HRES - 1), yc1 = min(max(y1, 0), HRES - 1);
            const int o00 = yc0 * WRES + xc0, o10 = yc0 * WRES + xc1;
            const int o01 = yc1 * WRES + xc0, o11 = yc1 * WRES + xc1;
            const float* base = fb + (size_t)pl * CH * HRES * WRES;
#pragma unroll 4
            for (int ch = 0; ch < CH; ++ch) {
                const float* f = base + (size_t)ch * HRES * WRES;
                acc[ch] += w00 * f[o00] + w10 * f[o10] + w01 * f[o01] + w11 * f[o11];
            }
        }
#pragma unroll
        for (int ch = 0; ch < CH; ++ch)
            sX[tid * CH + ch] = (_Float16)(acc[ch] * (1.f / 3.f));
    }
    __syncthreads();

    // ---- phase 2: WMMA MLP; each wave owns rows [wave*32, wave*32+32) ----

    // layer 0: [32 x 32] @ [32 x 64]^T  (K = 32 -> one WMMA step)
#pragma unroll
    for (int t = 0; t < 2; ++t) {
        const int rows = wave * 32 + t * 16;
        v16h a = loadA(sX, CH, rows, 0, lane);
#pragma unroll
        for (int c = 0; c < 4; ++c) {
            v8f d = {0.f, 0.f, 0.f, 0.f, 0.f, 0.f, 0.f, 0.f};
            d = wmma_f16(a, loadB(w0, CH, c * 16, 0, lane), d);
            store_act(d, sH1, HID, rows, c * 16, b0, true, lane);
        }
    }
    __syncthreads();

    // layer 1: [32 x 64] @ [64 x 64]^T  (K = 64 -> two WMMA steps)
#pragma unroll
    for (int t = 0; t < 2; ++t) {
        const int rows = wave * 32 + t * 16;
        v16h a0 = loadA(sH1, HID, rows, 0, lane);
        v16h a1 = loadA(sH1, HID, rows, 1, lane);
#pragma unroll
        for (int c = 0; c < 4; ++c) {
            v8f d = {0.f, 0.f, 0.f, 0.f, 0.f, 0.f, 0.f, 0.f};
            d = wmma_f16(a0, loadB(w1, HID, c * 16, 0, lane), d);
            d = wmma_f16(a1, loadB(w1, HID, c * 16, 1, lane), d);
            store_act(d, sH2, HID, rows, c * 16, b1, true, lane);
        }
    }
    __syncthreads();

    // layer 2: [32 x 64] @ [64 x 48]^T, linear + clamp, emit cols 0..32
#pragma unroll
    for (int t = 0; t < 2; ++t) {
        const int rows = wave * 32 + t * 16;
        v16h a0 = loadA(sH2, HID, rows, 0, lane);
        v16h a1 = loadA(sH2, HID, rows, 1, lane);
#pragma unroll
        for (int c = 0; c < 3; ++c) {
            v8f d = {0.f, 0.f, 0.f, 0.f, 0.f, 0.f, 0.f, 0.f};
            d = wmma_f16(a0, loadB(w2, HID, c * 16, 0, lane), d);
            d = wmma_f16(a1, loadB(w2, HID, c * 16, 1, lane), d);
            const int col = c * 16 + (lane & 15);
            if (col < OUTC) {
                const float bv = b2[col];
                const int r0 = rows + ((lane >> 4) << 3);
#pragma unroll
                for (int r = 0; r < 8; ++r) {
                    const int gm = m0 + r0 + r;
                    float v = d[r] + bv;
                    v = fminf(fmaxf(v, -CLAMPV), CLAMPV);
                    if (gm < M)
                        out[((size_t)n * M + gm) * OUTC + col] = v;
                }
            }
        }
    }
}

// ---------------------------------------------------------------------------
extern "C" void kernel_launch(void* const* d_in, const int* in_sizes, int n_in,
                              void* d_out, int out_size, void* d_ws, size_t ws_size,
                              hipStream_t stream)
{
    const float* planes = (const float*)d_in[0];
    const float* coords = (const float*)d_in[1];
    const float* wlat   = (const float*)d_in[2];
    const int*   b_idx  = (const int*)  d_in[3];
    const float* W0  = (const float*)d_in[4];
    const float* b0  = (const float*)d_in[5];
    const float* Ws0 = (const float*)d_in[6];
    const float* bs0 = (const float*)d_in[7];
    const float* W1  = (const float*)d_in[8];
    const float* b1  = (const float*)d_in[9];
    const float* Ws1 = (const float*)d_in[10];
    const float* bs1 = (const float*)d_in[11];
    const float* W2  = (const float*)d_in[12];
    const float* b2  = (const float*)d_in[13];
    const float* Ws2 = (const float*)d_in[14];
    const float* bs2 = (const float*)d_in[15];
    float* out = (float*)d_out;

    const int N = in_sizes[3];                 // 8
    const int M = in_sizes[1] / (3 * N);       // 65536

    _Float16* weff0 = (_Float16*)d_ws;
    _Float16* weff1 = weff0 + (size_t)N * HID  * CH;
    _Float16* weff2 = weff1 + (size_t)N * HID  * HID;

    modw_kernel<<<N, 64, 0, stream>>>(wlat, W0, Ws0, bs0, weff0, CH,  HID,  HID);
    modw_kernel<<<N, 64, 0, stream>>>(wlat, W1, Ws1, bs1, weff1, HID, HID,  HID);
    modw_kernel<<<N, 64, 0, stream>>>(wlat, W2, Ws2, bs2, weff2, HID, OUTC, OUTP);

    dim3 grid((M + PTS - 1) / PTS, N);
    triplane_mlp_kernel<<<grid, PTS, 0, stream>>>(planes, coords, b_idx,
                                                  weff0, weff1, weff2,
                                                  b0, b1, b2, out, M);
}
